// LocationAwareAttention_51144470560891
// MI455X (gfx1250) — matrix operand order repaired and checked by reading, not compiled, not run on hardware
//
#include <hip/hip_runtime.h>
#include <hip/hip_bf16.h>
#include <math.h>

// Problem constants (reference: B=32, L=4096, H=512)
#define B_ 32
#define L_ 4096
#define H_ 512
#define ROWS_PER_BLK 32
#define TILES_PER_B (L_ / ROWS_PER_BLK)   // 128
#define NT_TOTAL (H_ / 16)                // 32 column tiles of 16
#define KT_TOTAL (H_ / 32)                // 16 K-chunks of 32

typedef __bf16        v16bf __attribute__((ext_vector_type(16)));
typedef float         v8f   __attribute__((ext_vector_type(8)));
typedef float         f32x8 __attribute__((ext_vector_type(8)));
typedef unsigned int  u32x8 __attribute__((ext_vector_type(8)));

__device__ __forceinline__ unsigned short f2bf_rne(float f) {
    union { float f; unsigned u; } v; v.f = f;
    unsigned r = v.u + 0x7FFFu + ((v.u >> 16) & 1u);   // round-to-nearest-even
    return (unsigned short)(r >> 16);
}

// ---------------------------------------------------------------------------
// Kernel 1: pre[b,o] = sum_h query[b,h]*Wq[o,h] + bias[o] + conv_b[o]
// (folds the two broadcast bias terms into one per-(b,o) table)
// ---------------------------------------------------------------------------
__global__ __launch_bounds__(256)
void law_pre_kernel(const float* __restrict__ query, const float* __restrict__ Wq,
                    const float* __restrict__ bias, const float* __restrict__ conv_b,
                    float* __restrict__ pre) {
    int gid = blockIdx.x * 256 + threadIdx.x;      // 0 .. B*H-1
    int b = gid >> 9;                              // /512
    int o = gid & (H_ - 1);
    const float* q = query + (size_t)b * H_;
    const float* w = Wq + (size_t)o * H_;
    float s = 0.0f;
    #pragma unroll 4
    for (int h = 0; h < H_; ++h) s += q[h] * w[h];
    pre[gid] = s + bias[o] + conv_b[o];
}

// ---------------------------------------------------------------------------
// Kernel 2: pack Wv (H x H, row-major [o][h]) into bf16 WMMA B-fragments.
// B-matrix (32x16, K=h, N=o) per-lane layout used here (contiguous K, same as A):
//   lane = n + 16*khalf ; dword j holds K = khalf*16 + 2j (lo) , +1 (hi)
// Fragment dword index = ((kt*NT + nt)*32 + lane)*8 + j
// ---------------------------------------------------------------------------
__global__ __launch_bounds__(256)
void law_pack_wv_kernel(const float* __restrict__ Wv, unsigned* __restrict__ wvpack) {
    int gid = blockIdx.x * 256 + threadIdx.x;      // 0 .. KT*NT*32*8-1
    int j    = gid & 7;
    int lane = (gid >> 3) & 31;
    int nt   = (gid >> 8) & 31;
    int kt   = gid >> 13;
    int n  = nt * 16 + (lane & 15);
    int k0 = kt * 32 + (lane >> 4) * 16 + 2 * j;
    float e0 = Wv[(size_t)n * H_ + k0];
    float e1 = Wv[(size_t)n * H_ + k0 + 1];
    wvpack[gid] = (unsigned)f2bf_rne(e0) | ((unsigned)f2bf_rne(e1) << 16);
}

// ---------------------------------------------------------------------------
// Kernel 3: fused  v = value@Wv^T (bf16 WMMA, f32 acc)  ->  tanh(pre+v+conv)
//           -> score -> sigmoid -> per-block partial S and partial context.
// Block: 256 threads = 8 waves = 2 row-groups(16 rows) x 4 col-groups(128 cols)
// covering a 32x512 output tile for one (b, l-tile).
// ---------------------------------------------------------------------------
__global__ __launch_bounds__(256)
void law_main_kernel(const float* __restrict__ value,
                     const float* __restrict__ last_attn,
                     const float* __restrict__ conv_w,   // (H,1,3): conv_w[o*3+k]
                     const float* __restrict__ Ws,
                     const float* __restrict__ bs,       // scalar
                     const float* __restrict__ pre,      // B*H
                     const unsigned* __restrict__ wvpack,
                     float* __restrict__ sigbuf,         // B*L (unnormalized sigmoid)
                     float* __restrict__ part_S,         // B*TILES
                     float* __restrict__ part_ctx)       // B*TILES*H
{
    const int tid  = threadIdx.x;
    const int lane = tid & 31;
    const int wave = tid >> 5;
    const int mg   = wave >> 2;        // row group (16 rows each)
    const int ng   = wave & 3;         // col group (128 cols each)

    const int b     = blockIdx.x / TILES_PER_B;
    const int blk   = blockIdx.x % TILES_PER_B;
    const int lbase = blk * ROWS_PER_BLK;

    __shared__ float s_la[ROWS_PER_BLK + 2];   // last_attn[lbase-1 .. lbase+32]
    __shared__ float s_score[2][4][16];
    __shared__ float s_sig[ROWS_PER_BLK];

    if (tid < ROWS_PER_BLK + 2) {
        int idx = lbase - 1 + tid;
        s_la[tid] = (idx >= 0 && idx < L_) ? last_attn[(size_t)b * L_ + idx] : 0.0f;
    }

    const int half = lane >> 4;        // K-half for A, row-half for C/D
    const int lmod = lane & 15;
    const int arow = lbase + mg * 16 + lmod;

    // A-fragment base: two contiguous 8-float runs per K-chunk:
    //   elements 0..7  <- value[arow][kc*32 + half*16 + 0..7]
    //   elements 8..15 <- value[arow][kc*32 + half*16 + 8 + 0..7]
    const float* aptr = value + ((size_t)b * L_ + arow) * H_ + half * 16;

    v8f acc[8];
    #pragma unroll
    for (int t = 0; t < 8; ++t)
        #pragma unroll
        for (int i = 0; i < 8; ++i) acc[t][i] = 0.0f;

    for (int kc = 0; kc < KT_TOTAL; ++kc) {
        f32x8 r0 = *(const f32x8*)(aptr + kc * 32);
        f32x8 r1 = *(const f32x8*)(aptr + kc * 32 + 8);
        v16bf a;
        #pragma unroll
        for (int e = 0; e < 8; ++e) {
            a[e]     = (__bf16)r0[e];
            a[8 + e] = (__bf16)r1[e];
        }
        #pragma unroll
        for (int t = 0; t < 8; ++t) {
            int nt = ng * 8 + t;
            u32x8 raw = *(const u32x8*)(wvpack +
                         (((size_t)kc * NT_TOTAL + nt) * 32 + lane) * 8);
            v16bf bf = __builtin_bit_cast(v16bf, raw);
            acc[t] = __builtin_amdgcn_wmma_f32_16x16x32_bf16(
                         false, a, false, bf, (short)0, acc[t], false, false);
        }
    }

    __syncthreads();   // s_la visible to all waves

    // Epilogue: e = tanh(acc + pre[o] + conv(last_attn)) ; ps[r] += e*Ws[o]
    // C/D layout: reg r, lanes 0-15 -> (row r, col lane); lanes 16-31 -> (row 8+r)
    float ps[8];
    #pragma unroll
    for (int r = 0; r < 8; ++r) ps[r] = 0.0f;

    #pragma unroll
    for (int t = 0; t < 8; ++t) {
        int o = (ng * 8 + t) * 16 + lmod;
        float pre_o = pre[b * H_ + o];
        float w0 = conv_w[o * 3 + 0];
        float w1 = conv_w[o * 3 + 1];
        float w2 = conv_w[o * 3 + 2];
        float ws_o = Ws[o];
        #pragma unroll
        for (int r = 0; r < 8; ++r) {
            int ri = mg * 16 + r + 8 * half;   // row within 32-row block
            float x = acc[t][r] + pre_o
                    + w0 * s_la[ri] + w1 * s_la[ri + 1] + w2 * s_la[ri + 2];
            ps[r] += tanhf(x) * ws_o;
        }
    }

    // Reduce across the 16 lanes of each half (fixed order -> deterministic)
    #pragma unroll
    for (int r = 0; r < 8; ++r) {
        #pragma unroll
        for (int off = 1; off < 16; off <<= 1)
            ps[r] += __shfl_xor(ps[r], off, 32);
    }
    if (lmod == 0) {
        #pragma unroll
        for (int r = 0; r < 8; ++r)
            s_score[mg][ng][r + 8 * half] = ps[r];
    }
    __syncthreads();

    // Combine the 4 column-group partials (fixed order), sigmoid
    if (tid < ROWS_PER_BLK) {
        int rmg = tid >> 4, rr = tid & 15;
        float sc = s_score[rmg][0][rr] + s_score[rmg][1][rr]
                 + s_score[rmg][2][rr] + s_score[rmg][3][rr] + bs[0];
        float sg = 1.0f / (1.0f + expf(-sc));
        s_sig[tid] = sg;
        sigbuf[(size_t)b * L_ + lbase + tid] = sg;
    }
    __syncthreads();

    // Partial context: ctx[h] = sum_rows sig[row]*value[row][h]  (tile hot in L2)
    {
        int h = tid * 2;
        float c0 = 0.0f, c1 = 0.0f;
        const float* vp = value + ((size_t)b * L_ + lbase) * H_ + h;
        #pragma unroll 4
        for (int row = 0; row < ROWS_PER_BLK; ++row) {
            float sg = s_sig[row];
            c0 += sg * vp[0];
            c1 += sg * vp[1];
            vp += H_;
        }
        float* pc = part_ctx + ((size_t)b * TILES_PER_B + blk) * H_ + h;
        pc[0] = c0; pc[1] = c1;
    }
    if (tid == 0) {
        float s = 0.0f;
        for (int row = 0; row < ROWS_PER_BLK; ++row) s += s_sig[row];
        part_S[b * TILES_PER_B + blk] = s;
    }
}

// ---------------------------------------------------------------------------
// Kernel 4: deterministic final reduction + normalization + output assembly.
//   out[b, 0:H)   = (sum_blk part_ctx) / S
//   out[b, H:2H)  = query[b]
//   attn[b, l]    = sigbuf[b,l] / S
// ---------------------------------------------------------------------------
__global__ __launch_bounds__(256)
void law_finalize_kernel(const float* __restrict__ query,
                         const float* __restrict__ part_S,
                         const float* __restrict__ part_ctx,
                         const float* __restrict__ sigbuf,
                         float* __restrict__ out) {
    int b = blockIdx.x;
    int tid = threadIdx.x;
    __shared__ float sS;
    if (tid == 0) {
        float s = 0.0f;
        for (int i = 0; i < TILES_PER_B; ++i) s += part_S[b * TILES_PER_B + i];
        sS = s;
    }
    __syncthreads();
    float inv = 1.0f / sS;

    for (int h = tid; h < H_; h += 256) {
        float c = 0.0f;
        const float* pc = part_ctx + (size_t)b * TILES_PER_B * H_ + h;
        for (int i = 0; i < TILES_PER_B; ++i) c += pc[(size_t)i * H_];
        out[(size_t)b * 2 * H_ + h]      = c * inv;
        out[(size_t)b * 2 * H_ + H_ + h] = query[(size_t)b * H_ + h];
    }
    float* attn = out + (size_t)B_ * 2 * H_;
    for (int l = tid; l < L_; l += 256)
        attn[(size_t)b * L_ + l] = sigbuf[(size_t)b * L_ + l] * inv;
}

// ---------------------------------------------------------------------------
extern "C" void kernel_launch(void* const* d_in, const int* in_sizes, int n_in,
                              void* d_out, int out_size, void* d_ws, size_t ws_size,
                              hipStream_t stream) {
    const float* query     = (const float*)d_in[0];
    const float* value     = (const float*)d_in[1];
    const float* last_attn = (const float*)d_in[2];
    const float* conv_w    = (const float*)d_in[3];
    const float* conv_b    = (const float*)d_in[4];
    const float* Wq        = (const float*)d_in[5];
    const float* Wv        = (const float*)d_in[6];
    const float* Ws        = (const float*)d_in[7];
    const float* bs        = (const float*)d_in[8];
    const float* bias      = (const float*)d_in[9];

    char* p = (char*)d_ws;
    unsigned* wvpack = (unsigned*)p;  p += (size_t)KT_TOTAL * NT_TOTAL * 32 * 8 * sizeof(unsigned); // 512 KB
    float* pre       = (float*)p;     p += (size_t)B_ * H_ * sizeof(float);                          // 64 KB
    float* sigbuf    = (float*)p;     p += (size_t)B_ * L_ * sizeof(float);                          // 512 KB
    float* part_S    = (float*)p;     p += (size_t)B_ * TILES_PER_B * sizeof(float);                 // 16 KB
    float* part_ctx  = (float*)p;     /* B*TILES*H floats = 8 MB */

    law_pre_kernel<<<(B_ * H_) / 256, 256, 0, stream>>>(query, Wq, bias, conv_b, pre);
    law_pack_wv_kernel<<<(KT_TOTAL * NT_TOTAL * 32 * 8) / 256, 256, 0, stream>>>(Wv, wvpack);
    law_main_kernel<<<B_ * TILES_PER_B, 256, 0, stream>>>(
        value, last_attn, conv_w, Ws, bs, pre, wvpack, sigbuf, part_S, part_ctx);
    law_finalize_kernel<<<B_, 256, 0, stream>>>(query, part_S, part_ctx, sigbuf, (float*)d_out);
}
